// MultiHeadSelfAttention_68959994904666
// MI455X (gfx1250) — compile-verified, matrix operand order checked by reading, not run on hardware
//
#include <hip/hip_runtime.h>
#include <hip/hip_bf16.h>
#include <math.h>

// ---------------------------------------------------------------------------
// MHSA for MI455X (gfx1250, wave32, WMMA + TDM). B=2,S=2048,D=1024,H=16,dh=64.
//
//   0) one-shot fp32 -> bf16 conversion of x, w_qkv, w_out
//   1) QKV GEMM: 64x64 macro-tile/wave, 16 v_wmma_f32_16x16x32_bf16 per
//      K-step, b128 fragment loads, global_prefetch on the next K slice.
//      Epilogue: q,k -> (B,H,S,dh); V stored transposed -> (B,H,dh,S).
//   2) RoPE in place on bf16 q,k.
//   3) Flash attention: K/V tiles DMA'd into LDS by the Tensor Data Mover
//      (tensor_load_to_lds, double-buffered on TENSORcnt); scores stay in
//      registers/LDS; online softmax; P transposed via LDS bounce.
//   4) Output projection GEMM, fp32 result.
// ---------------------------------------------------------------------------

typedef __attribute__((ext_vector_type(16))) __bf16 v16bf;
typedef __attribute__((ext_vector_type(8)))  __bf16 v8bf;
typedef __attribute__((ext_vector_type(4)))  __bf16 v4bf;
typedef __attribute__((ext_vector_type(8)))  float  v8f;
typedef __attribute__((ext_vector_type(4)))  unsigned int u32x4;
typedef __attribute__((ext_vector_type(8)))  int i32x8;
typedef __attribute__((ext_vector_type(4)))  int i32x4;

#define B_  2
#define S_  2048
#define D_  1024
#define H_  16
#define DH_ 64

// 16x32 (rows x K) bf16 WMMA fragment from a K-contiguous bf16 matrix
// (global or LDS). ISA 7.12.2 16-bit A layout: lane = row (mod 16);
// lanes 0-15 hold K {0..7,16..23}, lanes 16-31 hold K {8..15,24..31}.
__device__ __forceinline__ v16bf load_frag_bf16(const __bf16* __restrict__ p,
                                                int ld, int lane) {
  const int r  = lane & 15;
  const int kb = (lane & 16) ? 8 : 0;
  const __bf16* q = p + (long)r * ld + kb;
  v8bf lo = *(const v8bf*)(q);
  v8bf hi = *(const v8bf*)(q + 16);
  return __builtin_shufflevector(lo, hi, 0, 1, 2, 3, 4, 5, 6, 7,
                                 8, 9, 10, 11, 12, 13, 14, 15);
}

__device__ __forceinline__ v8f wmma_bf16(v16bf a, v16bf b, v8f c) {
  return __builtin_amdgcn_wmma_f32_16x16x32_bf16(false, a, false, b,
                                                 (short)0, c, false, false);
}

// ---------------------------------------------------------------------------
// Tensor Data Mover: 2D bf16 tile load, global -> LDS (ISA 08 §8 D# layout).
// group0: [1:0]=count=1, [63:32]=lds_addr, [120:64]=global_addr, [127:126]=2.
// group1: [17:16]=data_size(1 -> 2B), [79:48]=tensor_dim0, [111:80]=tensor_dim1,
//         [127:112]=tile_dim0, [143:128]=tile_dim1, [207:160]=dim0_stride.
// This toolchain exposes the 6-arg builtin:
//   (u32x4 g0, i32x8 g1, i32x4 g2, i32x4 g3, i32x8 extra, i32 cpol)
// ---------------------------------------------------------------------------
__device__ __forceinline__ void tdm_load_2d_bf16(
    const __bf16* gsrc, unsigned lds_addr, unsigned tile_d0, unsigned tile_d1,
    unsigned tensor_d0, unsigned tensor_d1, unsigned long stride0) {
  const unsigned long long ga = (unsigned long long)gsrc;
  u32x4 g0;
  g0[0] = 1u;                                         // count=1, user mode
  g0[1] = lds_addr;                                   // LDS byte address
  g0[2] = (unsigned)(ga & 0xFFFFFFFFu);               // global_addr[31:0]
  g0[3] = (unsigned)((ga >> 32) & 0x01FFFFFFu) | (2u << 30);  // [56:32]|type=2
  i32x8 g1;
  g1[0] = (int)(1u << 16);                            // data_size=2B, mask=0
  g1[1] = (int)((tensor_d0 & 0xFFFFu) << 16);         // dim0[15:0] -> [63:48]
  g1[2] = (int)(((tensor_d0 >> 16) & 0xFFFFu) |
                ((tensor_d1 & 0xFFFFu) << 16));       // dim0[31:16], dim1[15:0]
  g1[3] = (int)(((tensor_d1 >> 16) & 0xFFFFu) |
                ((tile_d0 & 0xFFFFu) << 16));         // dim1[31:16], tile0
  g1[4] = (int)(tile_d1 & 0xFFFFu);                   // tile1; tile2=0
  g1[5] = (int)(stride0 & 0xFFFFFFFFu);               // dim0_stride[31:0]
  g1[6] = (int)((stride0 >> 32) & 0xFFFFu);           // dim0_stride[47:32]
  g1[7] = 0;                                          // dim1_stride unused (2D)
  const i32x4 z4 = {0, 0, 0, 0};
  const i32x8 z8 = {0, 0, 0, 0, 0, 0, 0, 0};
  __builtin_amdgcn_tensor_load_to_lds(g0, g1, z4, z4, z8, 0);
}

__device__ __forceinline__ unsigned lds_byte_addr(const void* p) {
  // Generic pointers to LDS carry the LDS byte offset in addr[31:0].
  return (unsigned)(unsigned long long)p;
}

// ---------------------------------------------------------------------------
// Kernel 0: fp32 -> bf16 (4 elements / thread; sizes are multiples of 1024).
// ---------------------------------------------------------------------------
__global__ __launch_bounds__(256) void cvt_bf16_kernel(
    const float* __restrict__ in, __bf16* __restrict__ out, long n) {
  long i = ((long)blockIdx.x * blockDim.x + threadIdx.x) * 4;
  if (i >= n) return;
  float4 v = *(const float4*)(in + i);
  v4bf o;
  o[0] = (__bf16)v.x; o[1] = (__bf16)v.y;
  o[2] = (__bf16)v.z; o[3] = (__bf16)v.w;
  *(v4bf*)(out + i) = o;
}

// ---------------------------------------------------------------------------
// Kernel 1: qkv = x @ w_qkv^T. M=4096, N=3072, K=1024. 64x64 tile per wave.
// ---------------------------------------------------------------------------
__global__ __launch_bounds__(128) void qkv_gemm_kernel(
    const __bf16* __restrict__ x, const __bf16* __restrict__ w,
    __bf16* __restrict__ q_ws, __bf16* __restrict__ k_ws,
    __bf16* __restrict__ vT_ws) {
  const int lane  = threadIdx.x & 31;
  const int wid   = threadIdx.x >> 5;
  const int mbase = blockIdx.x * 64;
  const int nbase = (blockIdx.y * 4 + wid) * 64;

  v8f acc[4][4] = {};
  for (int k = 0; k < D_; k += 32) {
    v16bf a[4], b[4];
#pragma unroll
    for (int i = 0; i < 4; ++i)
      a[i] = load_frag_bf16(x + (long)(mbase + 16 * i) * D_ + k, D_, lane);
#pragma unroll
    for (int j = 0; j < 4; ++j)
      b[j] = load_frag_bf16(w + (long)(nbase + 16 * j) * D_ + k, D_, lane);
    if (k + 32 < D_) {
      __builtin_prefetch(x + (long)mbase * D_ + k + 32, 0, 1);
      __builtin_prefetch(w + (long)nbase * D_ + k + 32, 0, 1);
    }
#pragma unroll
    for (int i = 0; i < 4; ++i)
#pragma unroll
      for (int j = 0; j < 4; ++j) acc[i][j] = wmma_bf16(a[i], b[j], acc[i][j]);
  }

  const int part   = nbase >> 10;               // 0=q 1=k 2=v
  const int h      = (nbase & (D_ - 1)) >> 6;   // head (64-wide stripe)
  const int n      = lane & 15;
  const int rowoff = (lane & 16) ? 8 : 0;
  __bf16* qk = (part == 0) ? q_ws : k_ws;
#pragma unroll
  for (int i = 0; i < 4; ++i) {
#pragma unroll
    for (int r = 0; r < 8; ++r) {
      const int m  = mbase + 16 * i + rowoff + r;  // m = b*S + s
      const int bb = m >> 11;
      const int s  = m & (S_ - 1);
      const long head = (long)bb * H_ + h;
#pragma unroll
      for (int j = 0; j < 4; ++j) {
        const int d = 16 * j + n;
        const __bf16 val = (__bf16)acc[i][j][r];
        if (part < 2) qk[(head * S_ + s) * DH_ + d] = val;
        else          vT_ws[(head * DH_ + d) * S_ + s] = val;  // transposed V
      }
    }
  }
}

// ---------------------------------------------------------------------------
// Kernel 2: RoPE in place on bf16 q,k.
// ---------------------------------------------------------------------------
__global__ __launch_bounds__(256) void rope_kernel(
    __bf16* __restrict__ q_ws, __bf16* __restrict__ k_ws,
    const int* __restrict__ pos) {
  const long total = (long)B_ * H_ * S_ * (DH_ / 2);
  long i = (long)blockIdx.x * blockDim.x + threadIdx.x;
  if (i >= total) return;
  const int  p   = (int)(i & (DH_ / 2 - 1));
  const long row = i >> 5;
  const int  s   = (int)(row & (S_ - 1));

  const float posv = (float)pos[s];
  const float freq = __powf(10000.0f, -2.0f * (float)p / (float)DH_);
  float sn, cs;
  __sincosf(posv * freq, &sn, &cs);

  const long idx = row * DH_ + 2 * p;
  float q0 = (float)q_ws[idx], q1 = (float)q_ws[idx + 1];
  q_ws[idx]     = (__bf16)(cs * q0 - sn * q1);
  q_ws[idx + 1] = (__bf16)(sn * q0 + cs * q1);
  float k0 = (float)k_ws[idx], k1 = (float)k_ws[idx + 1];
  k_ws[idx]     = (__bf16)(cs * k0 - sn * k1);
  k_ws[idx + 1] = (__bf16)(sn * k0 + cs * k1);
}

// ---------------------------------------------------------------------------
// Kernel 3: causal flash attention, one wave per (b,h, 16-row Q block).
// K/V tiles staged into LDS by the TDM, double-buffered on TENSORcnt.
// ---------------------------------------------------------------------------
__global__ __launch_bounds__(32) void flash_attn_kernel(
    const __bf16* __restrict__ q_ws, const __bf16* __restrict__ k_ws,
    const __bf16* __restrict__ vT_ws, __bf16* __restrict__ o_ws) {
  __shared__ __bf16 kbuf[2][32 * DH_];   // 32 keys x 64 dh   (8 KB)
  __shared__ __bf16 vbuf[2][DH_ * 32];   // 64 dh  x 32 keys  (8 KB)
  __shared__ __bf16 plds[16 * 32];       // P bounce: D-layout -> A-layout

  const int lane  = threadIdx.x & 31;
  const int qbase = blockIdx.x * 16;
  const int bh    = blockIdx.y;                // b*H + h
  const float scale = 0.125f;                  // 1/sqrt(64)

  const __bf16* qb = q_ws + ((long)bh * S_ + qbase) * DH_;
  const __bf16* kp = k_ws + (long)bh * S_ * DH_;
  const __bf16* vT = vT_ws + (long)bh * DH_ * S_;

  const v16bf qa0 = load_frag_bf16(qb, DH_, lane);
  const v16bf qa1 = load_frag_bf16(qb + 32, DH_, lane);

  const int n      = lane & 15;
  const int rowoff = (lane & 16) ? 8 : 0;

  float mrow[8], lrow[8];
  v8f o[4] = {v8f{}, v8f{}, v8f{}, v8f{}};
#pragma unroll
  for (int r = 0; r < 8; ++r) { mrow[r] = -1e30f; lrow[r] = 0.0f; }

  // TDM issue: K tile (32 rows x 64, row-major) and V^T tile (64 rows x 32).
  auto issue_tiles = [&](int buf, int kbase) {
    tdm_load_2d_bf16(kp + (long)kbase * DH_, lds_byte_addr(&kbuf[buf][0]),
                     /*tile*/ DH_, 32, /*tensor*/ DH_, S_ - kbase,
                     /*stride0*/ DH_);
    tdm_load_2d_bf16(vT + kbase, lds_byte_addr(&vbuf[buf][0]),
                     /*tile*/ 32, DH_, /*tensor*/ S_ - kbase, DH_,
                     /*stride0*/ S_);
  };

  const int nkb = ((qbase + 15) >> 5) + 1;     // causal frontier, 32-key blocks
  issue_tiles(0, 0);

  for (int kblk = 0; kblk < nkb; ++kblk) {
    const int kbase = kblk * 32;
    const int cur   = kblk & 1;
    const bool has_next = (kblk + 1) < nkb;
    if (has_next) {
      // Buffer cur^1 was last consumed at iteration kblk-1 (its LDS reads
      // already landed in VGPRs before those WMMAs executed) -> safe to DMA.
      issue_tiles(cur ^ 1, kbase + 32);
      __builtin_amdgcn_s_wait_tensorcnt(2);    // current block's 2 DMAs done
    } else {
      __builtin_amdgcn_s_wait_tensorcnt(0);
    }
    asm volatile("" ::: "memory");

    // --- S = Q K^T from LDS: 16x32 scores as two 16x16 tiles.
    const __bf16* kl = &kbuf[cur][0];
    v8f s0 = {}, s1 = {};
    s0 = wmma_bf16(qa0, load_frag_bf16(kl,                 DH_, lane), s0);
    s0 = wmma_bf16(qa1, load_frag_bf16(kl + 32,            DH_, lane), s0);
    s1 = wmma_bf16(qa0, load_frag_bf16(kl + 16 * DH_,      DH_, lane), s1);
    s1 = wmma_bf16(qa1, load_frag_bf16(kl + 16 * DH_ + 32, DH_, lane), s1);

#pragma unroll
    for (int r = 0; r < 8; ++r) { s0[r] *= scale; s1[r] *= scale; }

    if (kbase + 31 > qbase) {                  // causal mask near the diagonal
      const int qrow = qbase + rowoff;
#pragma unroll
      for (int r = 0; r < 8; ++r) {
        if (kbase + n      > qrow + r) s0[r] = -1e30f;
        if (kbase + 16 + n > qrow + r) s1[r] = -1e30f;
      }
    }

    // --- online softmax row statistics (16-lane half-group reductions).
    float a[8];
#pragma unroll
    for (int r = 0; r < 8; ++r) {
      float v = fmaxf(s0[r], s1[r]);
      v = fmaxf(v, __shfl_xor(v, 1, 32));
      v = fmaxf(v, __shfl_xor(v, 2, 32));
      v = fmaxf(v, __shfl_xor(v, 4, 32));
      v = fmaxf(v, __shfl_xor(v, 8, 32));
      const float mnew = fmaxf(mrow[r], v);
      a[r]    = __expf(mrow[r] - mnew);
      mrow[r] = mnew;
    }
#pragma unroll
    for (int r = 0; r < 8; ++r) {
      const float p0 = __expf(s0[r] - mrow[r]);
      const float p1 = __expf(s1[r] - mrow[r]);
      s0[r] = p0; s1[r] = p1;
      float t = p0 + p1;
      t += __shfl_xor(t, 1, 32);
      t += __shfl_xor(t, 2, 32);
      t += __shfl_xor(t, 4, 32);
      t += __shfl_xor(t, 8, 32);
      lrow[r] = lrow[r] * a[r] + t;
      o[0][r] *= a[r]; o[1][r] *= a[r]; o[2][r] *= a[r]; o[3][r] *= a[r];
    }

    // --- P: D-layout -> A-fragment layout via LDS (wave-local, in-order LDS).
#pragma unroll
    for (int r = 0; r < 8; ++r) {
      const int m = rowoff + r;
      plds[m * 32 + n]      = (__bf16)s0[r];
      plds[m * 32 + 16 + n] = (__bf16)s1[r];
    }
    __builtin_amdgcn_wave_barrier();
    asm volatile("" ::: "memory");
    v16bf pa = load_frag_bf16(plds, 32, lane);
    asm volatile("" ::: "memory");
    __builtin_amdgcn_wave_barrier();

    // --- O += P V from LDS (V^T tile: row = dh, K-contiguous keys).
#pragma unroll
    for (int t = 0; t < 4; ++t)
      o[t] = wmma_bf16(pa, load_frag_bf16(&vbuf[cur][t * 16 * 32], 32, lane),
                       o[t]);
  }

  // Normalize; store bf16 as (B, S, D) for the out-projection GEMM.
  const int bb = bh >> 4;
  const int h  = bh & (H_ - 1);
#pragma unroll
  for (int r = 0; r < 8; ++r) {
    const float inv = 1.0f / lrow[r];
    const int  s    = qbase + rowoff + r;
    const long base = ((long)bb * S_ + s) * D_ + h * DH_ + n;
#pragma unroll
    for (int t = 0; t < 4; ++t)
      o_ws[base + t * 16] = (__bf16)(o[t][r] * inv);
  }
}

// ---------------------------------------------------------------------------
// Kernel 4: out = attn_out @ w_out^T. M=4096, N=1024, K=1024. fp32 out.
// ---------------------------------------------------------------------------
__global__ __launch_bounds__(128) void out_gemm_kernel(
    const __bf16* __restrict__ a_in, const __bf16* __restrict__ w,
    float* __restrict__ out) {
  const int lane  = threadIdx.x & 31;
  const int wid   = threadIdx.x >> 5;
  const int mbase = blockIdx.x * 64;
  const int nbase = (blockIdx.y * 4 + wid) * 64;

  v8f acc[4][4] = {};
  for (int k = 0; k < D_; k += 32) {
    v16bf a[4], b[4];
#pragma unroll
    for (int i = 0; i < 4; ++i)
      a[i] = load_frag_bf16(a_in + (long)(mbase + 16 * i) * D_ + k, D_, lane);
#pragma unroll
    for (int j = 0; j < 4; ++j)
      b[j] = load_frag_bf16(w + (long)(nbase + 16 * j) * D_ + k, D_, lane);
    if (k + 32 < D_) {
      __builtin_prefetch(a_in + (long)mbase * D_ + k + 32, 0, 1);
      __builtin_prefetch(w + (long)nbase * D_ + k + 32, 0, 1);
    }
#pragma unroll
    for (int i = 0; i < 4; ++i)
#pragma unroll
      for (int j = 0; j < 4; ++j) acc[i][j] = wmma_bf16(a[i], b[j], acc[i][j]);
  }

  const int n      = lane & 15;
  const int rowoff = (lane & 16) ? 8 : 0;
#pragma unroll
  for (int i = 0; i < 4; ++i)
#pragma unroll
    for (int r = 0; r < 8; ++r) {
      const long m = mbase + 16 * i + rowoff + r;
#pragma unroll
      for (int j = 0; j < 4; ++j)
        out[m * D_ + nbase + 16 * j + n] = acc[i][j][r];
    }
}

// ---------------------------------------------------------------------------
extern "C" void kernel_launch(void* const* d_in, const int* in_sizes, int n_in,
                              void* d_out, int out_size, void* d_ws,
                              size_t ws_size, hipStream_t stream) {
  const float* x     = (const float*)d_in[0];  // (B,S,D)
  const float* w_qkv = (const float*)d_in[1];  // (3D,D)
  const float* w_out = (const float*)d_in[2];  // (D,D)
  const int*   pos   = (const int*)d_in[3];    // (S)
  float* out = (float*)d_out;                  // (B,S,D)

  const long nX  = (long)B_ * S_ * D_;
  const long nWq = (long)3 * D_ * D_;
  const long nWo = (long)D_ * D_;
  const long nH  = (long)B_ * H_ * S_ * DH_;

  __bf16* x_bf  = (__bf16*)d_ws;
  __bf16* wq_bf = x_bf + nX;
  __bf16* wo_bf = wq_bf + nWq;
  __bf16* q_bf  = wo_bf + nWo;
  __bf16* k_bf  = q_bf + nH;
  __bf16* vT_bf = k_bf + nH;
  __bf16* o_bf  = vT_bf + nH;                  // total ~50 MB

  cvt_bf16_kernel<<<(int)(nX / 4 / 256), 256, 0, stream>>>(x, x_bf, nX);
  cvt_bf16_kernel<<<(int)(nWq / 4 / 256), 256, 0, stream>>>(w_qkv, wq_bf, nWq);
  cvt_bf16_kernel<<<(int)(nWo / 4 / 256), 256, 0, stream>>>(w_out, wo_bf, nWo);

  qkv_gemm_kernel<<<dim3(64, 12), 128, 0, stream>>>(x_bf, wq_bf, q_bf, k_bf,
                                                    vT_bf);
  {
    const long total = (long)B_ * H_ * S_ * (DH_ / 2);
    rope_kernel<<<(int)((total + 255) / 256), 256, 0, stream>>>(q_bf, k_bf,
                                                                pos);
  }
  flash_attn_kernel<<<dim3(S_ / 16, B_ * H_), 32, 0, stream>>>(q_bf, k_bf,
                                                               vT_bf, o_bf);
  out_gemm_kernel<<<dim3(64, 4), 128, 0, stream>>>(o_bf, wo_bf, out);
}
